// SpatialGAT_84172769067900
// MI455X (gfx1250) — matrix operand order repaired and checked by reading, not compiled, not run on hardware
//
#include <hip/hip_runtime.h>
#include <hip/hip_bf16.h>

#define NN    20000
#define EE    320000
#define EAE   340000   // EE + NN (with self loops)
#define FIN   256
#define HIDC  32
#define HEADS 8
#define HC1   256      // HEADS*HIDC
#define ED    8

typedef _Float16 h8v  __attribute__((ext_vector_type(8)));
typedef _Float16 h16v __attribute__((ext_vector_type(16)));
typedef float    f8v  __attribute__((ext_vector_type(8)));

__device__ __forceinline__ void atomicMaxF(float* addr, float val) {
  if (val >= 0.0f) atomicMax((int*)addr, __float_as_int(val));
  else             atomicMin((unsigned int*)addr, __float_as_uint(val));
}

// ---------------- WMMA GEMM: C[M,Nout] = A[M,K] (f16) * B[K,Nout], B supplied transposed Bt[Nout,K]
// One wave per 16x16 output tile. Fragment k order used here (contiguous, same for A and B):
//  A (16x32 f16): lanes 0-15 row M=lane hold K 0..15, lanes 16-31 hold K 16..31
//  B (32x16 f16): lanes 0-15 col N=lane hold K 0..15, lanes 16-31 hold K 16..31
template<int K>
__global__ void k_wmma_gemm(const _Float16* __restrict__ A,
                            const _Float16* __restrict__ Bt,
                            float* __restrict__ C,
                            int ntiles, int Nout) {
  const int lane = threadIdx.x & 31;
  const int tile = blockIdx.x * (blockDim.x >> 5) + (threadIdx.x >> 5);
  const int m0  = (tile / ntiles) << 4;
  const int n0  = (tile % ntiles) << 4;
  const int idx = lane & 15;
  const int hi  = lane >> 4;

  const _Float16* ap = A  + (size_t)(m0 + idx) * K + hi * 16;
  const _Float16* bp = Bt + (size_t)(n0 + idx) * K + hi * 16;

  f8v acc = {};
#pragma unroll
  for (int k = 0; k < K; k += 32) {
    h16v a, b;
    a = *(const h16v*)(ap + k);
    b = *(const h16v*)(bp + k);
    acc = __builtin_amdgcn_wmma_f32_16x16x32_f16(false, a, false, b,
                                                 (short)0, acc, false, false);
  }
  const int row = m0 + hi * 8;
  const int col = n0 + idx;
#pragma unroll
  for (int i = 0; i < 8; i++) C[(size_t)(row + i) * Nout + col] = acc[i];
}

// ---------------- small helpers
__global__ void k_fill(float* __restrict__ p, float v, int n) {
  int t = blockIdx.x * blockDim.x + threadIdx.x;
  if (t < n) p[t] = v;
}

__global__ void k_f32_to_f16(const float* __restrict__ in, _Float16* __restrict__ o, int n) {
  int t = blockIdx.x * blockDim.x + threadIdx.x;
  if (t < n) o[t] = (_Float16)in[t];
}

__global__ void k_relu_f16(const float* __restrict__ in, _Float16* __restrict__ o, int n) {
  int t = blockIdx.x * blockDim.x + threadIdx.x;
  if (t < n) o[t] = (_Float16)fmaxf(in[t], 0.0f);
}

__global__ void k_transpose_f16(const float* __restrict__ W, _Float16* __restrict__ Wt,
                                int K, int Nout) {
  int t = blockIdx.x * blockDim.x + threadIdx.x;
  if (t >= K * Nout) return;
  int k = t / Nout, n = t % Nout;
  Wt[(size_t)n * K + k] = (_Float16)W[t];
}

__global__ void k_bias_init(float* __restrict__ o, const float* __restrict__ b, int n, int cols) {
  int t = blockIdx.x * blockDim.x + threadIdx.x;
  if (t < n) o[t] = b[t % cols];
}

// ---------------- edge embedding: e = relu(attr @ Wse + bse); also degree + scatter-sum into tail
__global__ void k_edge_embed(const float* __restrict__ attr, const int* __restrict__ dst,
                             const float* __restrict__ Wse, const float* __restrict__ bse,
                             float* __restrict__ e_all, float* __restrict__ deg, int E_) {
  int e = blockIdx.x * blockDim.x + threadIdx.x;
  if (e >= E_) return;
  float a0 = attr[2 * e], a1 = attr[2 * e + 1];
  int d = dst[e];
  atomicAdd(&deg[d], 1.0f);
#pragma unroll
  for (int j = 0; j < ED; j++) {
    float v = fmaxf(a0 * Wse[j] + a1 * Wse[ED + j] + bse[j], 0.0f);
    e_all[(size_t)e * ED + j] = v;
    atomicAdd(&e_all[((size_t)E_ + d) * ED + j], v);
  }
}

__global__ void k_edge_mean(float* __restrict__ tail, const float* __restrict__ deg, int n) {
  int t = blockIdx.x * blockDim.x + threadIdx.x;
  if (t >= n) return;
  float s = 1.0f / fmaxf(deg[t], 1.0f);
#pragma unroll
  for (int j = 0; j < ED; j++) tail[(size_t)t * ED + j] *= s;
}

// M[k,h] = sum_c We[k, h*C + c] * ae[h, c]   (collapses a_edge = (e@We)·ae -> e@M)
__global__ void k_edge_att_mat(const float* __restrict__ We, const float* __restrict__ ae,
                               float* __restrict__ M, int H, int C) {
  int t = threadIdx.x;
  if (t >= ED * H) return;
  int k = t / H, h = t % H;
  float s = 0.f;
  for (int c = 0; c < C; c++) s += We[(size_t)k * H * C + h * C + c] * ae[h * C + c];
  M[k * H + h] = s;
}

// a_src[n,h], a_dst[n,h] = <h[n,h,:], att>
__global__ void k_node_att(const float* __restrict__ hmat, const float* __restrict__ att_s,
                           const float* __restrict__ att_d,
                           float* __restrict__ aS, float* __restrict__ aD,
                           int nNodes, int H, int C) {
  int t = blockIdx.x * blockDim.x + threadIdx.x;
  if (t >= nNodes * H) return;
  int n = t / H, h = t % H;
  const float* row = hmat + (size_t)n * H * C + h * C;
  float ss = 0.f, sd = 0.f;
  for (int c = 0; c < C; c++) { ss += row[c] * att_s[h * C + c]; sd += row[c] * att_d[h * C + c]; }
  aS[t] = ss; aD[t] = sd;
}

// logits + segment max  (edges id<E_ real, id>=E_ self loops)
template<int H>
__global__ void k_att_logits(const int* __restrict__ src, const int* __restrict__ dst,
                             const float* __restrict__ e_all, const float* __restrict__ M,
                             const float* __restrict__ aS, const float* __restrict__ aD,
                             float* __restrict__ logit, float* __restrict__ amax, int E_, int EA_) {
  int t = blockIdx.x * blockDim.x + threadIdx.x;
  if (t >= EA_ * H) return;
  int id = t / H, h = t % H;
  int s = (id < E_) ? src[id] : id - E_;
  int d = (id < E_) ? dst[id] : id - E_;
  float ae = 0.f;
  const float* er = e_all + (size_t)id * ED;
#pragma unroll
  for (int k = 0; k < ED; k++) ae += er[k] * M[k * H + h];
  float x = aS[s * H + h] + aD[d * H + h] + ae;
  x = (x > 0.f) ? x : 0.2f * x;     // leaky_relu 0.2
  logit[t] = x;
  atomicMaxF(&amax[d * H + h], x);
}

template<int H>
__global__ void k_att_exp(const int* __restrict__ dst, float* __restrict__ alpha,
                          const float* __restrict__ amax, float* __restrict__ denom,
                          int E_, int EA_) {
  int t = blockIdx.x * blockDim.x + threadIdx.x;
  if (t >= EA_ * H) return;
  int id = t / H, h = t % H;
  int d = (id < E_) ? dst[id] : id - E_;
  float v = __expf(alpha[t] - amax[d * H + h]);
  alpha[t] = v;
  atomicAdd(&denom[d * H + h], v);
}

// out[dst, h, c] += alpha/denom * feat[src, h, c]   (one thread per (edge, h*C+c))
template<int H, int C>
__global__ void k_att_agg(const int* __restrict__ src, const int* __restrict__ dst,
                          const float* __restrict__ alpha, const float* __restrict__ denom,
                          const float* __restrict__ feat, float* __restrict__ outp,
                          int E_, int EA_) {
  int t = blockIdx.x * blockDim.x + threadIdx.x;
  if (t >= EA_ * H * C) return;
  int id = t / (H * C);
  int j  = t % (H * C);
  int h  = j / C;
  int s = (id < E_) ? src[id] : id - E_;
  int d = (id < E_) ? dst[id] : id - E_;
  float w = alpha[(size_t)id * H + h] / (denom[d * H + h] + 1e-16f);
  atomicAdd(&outp[(size_t)d * H * C + j], w * feat[(size_t)s * H * C + j]);
}

extern "C" void kernel_launch(void* const* d_in, const int* in_sizes, int n_in,
                              void* d_out, int out_size, void* d_ws, size_t ws_size,
                              hipStream_t stream) {
  (void)in_sizes; (void)n_in; (void)out_size; (void)ws_size;
  const float* x    = (const float*)d_in[0];
  const int*   ei   = (const int*)d_in[1];
  const float* eatt = (const float*)d_in[2];
  const float* Wse  = (const float*)d_in[3];
  const float* bse  = (const float*)d_in[4];
  const float* W1   = (const float*)d_in[5];
  const float* as1  = (const float*)d_in[6];
  const float* ad1  = (const float*)d_in[7];
  const float* We1  = (const float*)d_in[8];
  const float* ae1  = (const float*)d_in[9];
  const float* b1   = (const float*)d_in[10];
  const float* W2   = (const float*)d_in[11];
  const float* as2  = (const float*)d_in[12];
  const float* ad2  = (const float*)d_in[13];
  const float* We2  = (const float*)d_in[14];
  const float* ae2  = (const float*)d_in[15];
  const float* b2   = (const float*)d_in[16];
  const int* src = ei;
  const int* dst = ei + EE;
  float* outF = (float*)d_out;

  char* p = (char*)d_ws;
  auto alloc = [&](size_t bytes) -> char* {
    char* r = p; p += (bytes + 255) & ~(size_t)255; return r;
  };
  float*    e_all = (float*)alloc((size_t)EAE * ED * 4);
  float*    deg   = (float*)alloc((size_t)NN * 4);
  _Float16* x16   = (_Float16*)alloc((size_t)NN * FIN * 2);
  _Float16* W1t   = (_Float16*)alloc((size_t)FIN * HC1 * 2);
  _Float16* W2t   = (_Float16*)alloc((size_t)HC1 * HIDC * 2);
  float*    h1    = (float*)alloc((size_t)NN * HC1 * 4);
  float*    aS1   = (float*)alloc((size_t)NN * HEADS * 4);
  float*    aD1   = (float*)alloc((size_t)NN * HEADS * 4);
  float*    M1    = (float*)alloc(ED * HEADS * 4);
  float*    M2    = (float*)alloc(ED * 4);
  float*    ab1   = (float*)alloc((size_t)EAE * HEADS * 4);
  float*    mx1   = (float*)alloc((size_t)NN * HEADS * 4);
  float*    dn1   = (float*)alloc((size_t)NN * HEADS * 4);
  float*    out1  = (float*)alloc((size_t)NN * HC1 * 4);
  _Float16* hr16  = (_Float16*)alloc((size_t)NN * HC1 * 2);
  float*    h2    = (float*)alloc((size_t)NN * HIDC * 4);
  float*    aS2   = (float*)alloc((size_t)NN * 4);
  float*    aD2   = (float*)alloc((size_t)NN * 4);
  float*    ab2   = (float*)alloc((size_t)EAE * 4);
  float*    mx2   = (float*)alloc((size_t)NN * 4);
  float*    dn2   = (float*)alloc((size_t)NN * 4);

  const int B = 256;
  auto G = [](long n, int b) { return (int)((n + b - 1) / b); };
  const float NEGINF = -3.402823466e38f;

  // init (must run every call: atomics accumulate)
  k_fill<<<G(NN, B), B, 0, stream>>>(deg, 0.f, NN);
  k_fill<<<G((long)NN * ED, B), B, 0, stream>>>(e_all + (size_t)EE * ED, 0.f, NN * ED);
  k_fill<<<G((long)NN * HEADS, B), B, 0, stream>>>(dn1, 0.f, NN * HEADS);
  k_fill<<<G(NN, B), B, 0, stream>>>(dn2, 0.f, NN);
  k_fill<<<G((long)NN * HEADS, B), B, 0, stream>>>(mx1, NEGINF, NN * HEADS);
  k_fill<<<G(NN, B), B, 0, stream>>>(mx2, NEGINF, NN);

  // fp16 staging for WMMA (weights pre-transposed -> contiguous B fragments)
  k_f32_to_f16<<<G((long)NN * FIN, B), B, 0, stream>>>(x, x16, NN * FIN);
  k_transpose_f16<<<G((long)FIN * HC1, B), B, 0, stream>>>(W1, W1t, FIN, HC1);
  k_transpose_f16<<<G((long)HC1 * HIDC, B), B, 0, stream>>>(W2, W2t, HC1, HIDC);

  // edge embedding, degree, self-loop mean attr, collapsed edge-attention matrices
  k_edge_embed<<<G(EE, B), B, 0, stream>>>(eatt, dst, Wse, bse, e_all, deg, EE);
  k_edge_mean<<<G(NN, B), B, 0, stream>>>(e_all + (size_t)EE * ED, deg, NN);
  k_edge_att_mat<<<1, 64, 0, stream>>>(We1, ae1, M1, HEADS, HIDC);
  k_edge_att_mat<<<1, 8, 0, stream>>>(We2, ae2, M2, 1, HIDC);

  // ---- conv1
  k_wmma_gemm<FIN><<<(NN / 16) * (HC1 / 16) / 4, 128, 0, stream>>>(x16, W1t, h1, HC1 / 16, HC1);
  k_node_att<<<G((long)NN * HEADS, B), B, 0, stream>>>(h1, as1, ad1, aS1, aD1, NN, HEADS, HIDC);
  k_att_logits<HEADS><<<G((long)EAE * HEADS, B), B, 0, stream>>>(src, dst, e_all, M1, aS1, aD1,
                                                                 ab1, mx1, EE, EAE);
  k_att_exp<HEADS><<<G((long)EAE * HEADS, B), B, 0, stream>>>(dst, ab1, mx1, dn1, EE, EAE);
  k_bias_init<<<G((long)NN * HC1, B), B, 0, stream>>>(out1, b1, NN * HC1, HC1);
  k_att_agg<HEADS, HIDC><<<G((long)EAE * HC1, B), B, 0, stream>>>(src, dst, ab1, dn1, h1, out1,
                                                                  EE, EAE);
  k_relu_f16<<<G((long)NN * HC1, B), B, 0, stream>>>(out1, hr16, NN * HC1);

  // ---- conv2 (H=1)
  k_wmma_gemm<HC1><<<(NN / 16) * (HIDC / 16) / 4, 128, 0, stream>>>(hr16, W2t, h2, HIDC / 16, HIDC);
  k_node_att<<<G(NN, B), B, 0, stream>>>(h2, as2, ad2, aS2, aD2, NN, 1, HIDC);
  k_att_logits<1><<<G(EAE, B), B, 0, stream>>>(src, dst, e_all, M2, aS2, aD2, ab2, mx2, EE, EAE);
  k_att_exp<1><<<G(EAE, B), B, 0, stream>>>(dst, ab2, mx2, dn2, EE, EAE);
  k_bias_init<<<G((long)NN * HIDC, B), B, 0, stream>>>(outF, b2, NN * HIDC, HIDC);
  k_att_agg<1, HIDC><<<G((long)EAE * HIDC, B), B, 0, stream>>>(src, dst, ab2, dn2, h2, outF,
                                                               EE, EAE);
}
